// AttnBlock_87892210745813
// MI455X (gfx1250) — matrix operand order repaired and checked by reading, not compiled, not run on hardware
//
#include <hip/hip_runtime.h>
#include <stdint.h>

// ---------------------------------------------------------------------------
// Problem constants (from the reference): x:[2,2048,1024], 16 heads x 64
// ---------------------------------------------------------------------------
#define EMB   1024
#define HEADS 16
#define HD    64
#define BATCH 2
#define SEQ   2048
#define MTOT  (BATCH * SEQ)          // 4096 rows

typedef __attribute__((ext_vector_type(16))) _Float16 v16h;
typedef __attribute__((ext_vector_type(8)))  float    v8f;
typedef __attribute__((ext_vector_type(4)))  int      v4i;

union FragU { uint4 u[2]; v16h h; };

// A-matrix fragment (16x32 f16, MxK). This code's contiguous k order (B matches):
// lane&15 = M row; lanes<16 hold K = 0..15 ; lanes>=16 hold K = 16..31
static __device__ inline v16h load_fragA(const _Float16* p, int sel) {
    FragU f;
    f.u[0] = *(const uint4*)(p + sel * 16);       // K: sel*16 .. sel*16+7
    f.u[1] = *(const uint4*)(p + sel * 16 + 8);   // K: sel*16+8 .. sel*16+15
    return f.h;
}

// B-matrix fragment (32x16 f16, KxN). lane&15 = N col; per lane the 16 K
// values are contiguous starting at sel*16. p = col base (k contiguous).
static __device__ inline v16h load_fragB(const _Float16* p, int sel) {
    const uint4* q = (const uint4*)(p + sel * 16);
    FragU f;
    f.u[0] = q[0];                                 // K: sel*16 .. +7
    f.u[1] = q[1];                                 // K: sel*16+8 .. +15
    return f.h;
}

#define WMMA_F16(a, b, c) \
    __builtin_amdgcn_wmma_f32_16x16x32_f16(false, (a), false, (b), (short)0, (c), false, false)

// ---------------------------------------------------------------------------
// CDNA5 async global->LDS staging (ASYNCcnt-tracked). The builtin prototype
// (from the compiler diagnostic) takes int4 pointers: (int4 AS1*, int4 AS3*,
// int offset, int cpol). Guarded to degrade to a plain copy if absent.
// ---------------------------------------------------------------------------
#if __has_builtin(__builtin_amdgcn_global_load_async_to_lds_b128)
#define HAS_ASYNC_LDS 1
#else
#define HAS_ASYNC_LDS 0
#endif

typedef __attribute__((address_space(1))) v4i glob_v4i;
typedef __attribute__((address_space(3))) v4i lds_v4i;

static __device__ inline void async_copy_b128(const _Float16* g, _Float16* l) {
#if HAS_ASYNC_LDS
    __builtin_amdgcn_global_load_async_to_lds_b128((glob_v4i*)g, (lds_v4i*)l, 0, 0);
#else
    *(uint4*)l = *(const uint4*)g;
#endif
}

static __device__ inline void wait_async_lds() {
#if HAS_ASYNC_LDS
#if __has_builtin(__builtin_amdgcn_s_wait_asynccnt)
    __builtin_amdgcn_s_wait_asynccnt(0);
#else
    asm volatile("s_wait_asynccnt 0x0" ::: "memory");
#endif
#endif
}

// ---------------------------------------------------------------------------
// fp32 -> f16 cast (weights), grid-stride
// ---------------------------------------------------------------------------
__global__ __launch_bounds__(256) void cast_f16_kernel(const float* __restrict__ src,
                                                       _Float16* __restrict__ dst, int n) {
    for (int i = blockIdx.x * blockDim.x + threadIdx.x; i < n; i += gridDim.x * blockDim.x)
        dst[i] = (_Float16)src[i];
}

// ---------------------------------------------------------------------------
// LayerNorm: one row (1024 floats) per block of 256 threads, f16 output
// ---------------------------------------------------------------------------
__global__ __launch_bounds__(256) void ln_kernel(const float* __restrict__ x,
                                                 const float* __restrict__ gamma,
                                                 const float* __restrict__ beta,
                                                 _Float16* __restrict__ h) {
    const int row = blockIdx.x;
    const int t   = threadIdx.x;
    const float* xr = x + (size_t)row * EMB;
    float4 v = ((const float4*)xr)[t];

    float s  = v.x + v.y + v.z + v.w;
    float s2 = v.x * v.x + v.y * v.y + v.z * v.z + v.w * v.w;
    for (int m = 16; m >= 1; m >>= 1) {
        s  += __shfl_xor(s,  m, 32);
        s2 += __shfl_xor(s2, m, 32);
    }
    __shared__ float ps[8], ps2[8];
    if ((t & 31) == 0) { ps[t >> 5] = s; ps2[t >> 5] = s2; }
    __syncthreads();
    float S = 0.f, S2 = 0.f;
#pragma unroll
    for (int i = 0; i < 8; ++i) { S += ps[i]; S2 += ps2[i]; }

    const float mu  = S * (1.0f / EMB);
    const float var = S2 * (1.0f / EMB) - mu * mu;
    const float rs  = rsqrtf(var + 1e-5f);

    _Float16* hr = h + (size_t)row * EMB;
    const int c = t * 4;
    hr[c + 0] = (_Float16)((v.x - mu) * rs * gamma[c + 0] + beta[c + 0]);
    hr[c + 1] = (_Float16)((v.y - mu) * rs * gamma[c + 1] + beta[c + 1]);
    hr[c + 2] = (_Float16)((v.z - mu) * rs * gamma[c + 2] + beta[c + 2]);
    hr[c + 3] = (_Float16)((v.w - mu) * rs * gamma[c + 3] + beta[c + 3]);
}

// ---------------------------------------------------------------------------
// GEMM core: Out[M,N] = A[M,K] * W[N,K]^T (+bias).  f16 in, fp32 accum.
// Block = 4 waves (128 thr). Block tile 128(M) x 64(N); wave tile 32 x 64.
// Weight k-slices (64 cols x 32 k) staged in double-buffered LDS via
// GLOBAL_LOAD_ASYNC_TO_LDS_B128, shared by all 4 waves.
// grid = (N/64, M/128)
// ---------------------------------------------------------------------------
#define GEMM_BODY(A_, W_, n0_, m0_)                                            \
    __shared__ _Float16 ldsB[2][64 * 32]; /* 2 x 4 KiB */                      \
    const int lane = threadIdx.x & 31;                                         \
    const int l15  = lane & 15;                                                \
    const int sel  = lane >> 4;                                                \
    const int wv   = threadIdx.x >> 5;                                         \
    const int m0_w = (m0_) + wv * 32;                                          \
    /* staging map: 256 b128 chunks per buffer, 2 per thread */                \
    const int ci0 = threadIdx.x * 2;                                           \
    const int sn0 = ci0 >> 2, sk0 = (ci0 & 3) * 8;                             \
    const int sn1 = (ci0 + 1) >> 2, sk1 = ((ci0 + 1) & 3) * 8;                 \
    const _Float16* wsrc0 = (W_) + (size_t)((n0_) + sn0) * EMB + sk0;          \
    const _Float16* wsrc1 = (W_) + (size_t)((n0_) + sn1) * EMB + sk1;          \
    const _Float16* arowL = (A_) + (size_t)(m0_w + l15) * EMB;                 \
    const _Float16* arowH = (A_) + (size_t)(m0_w + 16 + l15) * EMB;            \
    v8f accL0 = {}, accL1 = {}, accL2 = {}, accL3 = {};                        \
    v8f accH0 = {}, accH1 = {}, accH2 = {}, accH3 = {};                        \
    int p = 0;                                                                 \
    async_copy_b128(wsrc0, &ldsB[0][sn0 * 32 + sk0]);                          \
    async_copy_b128(wsrc1, &ldsB[0][sn1 * 32 + sk1]);                          \
    for (int k0 = 0; k0 < EMB; k0 += 32) {                                     \
        wait_async_lds();                                                      \
        __syncthreads();                                                       \
        if (k0 + 32 < EMB) {                                                   \
            async_copy_b128(wsrc0 + k0 + 32, &ldsB[p ^ 1][sn0 * 32 + sk0]);    \
            async_copy_b128(wsrc1 + k0 + 32, &ldsB[p ^ 1][sn1 * 32 + sk1]);    \
        }                                                                      \
        v16h aL = load_fragA(arowL + k0, sel);                                 \
        v16h aH = load_fragA(arowH + k0, sel);                                 \
        v16h b0 = load_fragB(&ldsB[p][( 0 + l15) * 32], sel);                  \
        v16h b1 = load_fragB(&ldsB[p][(16 + l15) * 32], sel);                  \
        v16h b2 = load_fragB(&ldsB[p][(32 + l15) * 32], sel);                  \
        v16h b3 = load_fragB(&ldsB[p][(48 + l15) * 32], sel);                  \
        accL0 = WMMA_F16(aL, b0, accL0);                                       \
        accL1 = WMMA_F16(aL, b1, accL1);                                       \
        accL2 = WMMA_F16(aL, b2, accL2);                                       \
        accL3 = WMMA_F16(aL, b3, accL3);                                       \
        accH0 = WMMA_F16(aH, b0, accH0);                                       \
        accH1 = WMMA_F16(aH, b1, accH1);                                       \
        accH2 = WMMA_F16(aH, b2, accH2);                                       \
        accH3 = WMMA_F16(aH, b3, accH3);                                       \
        __syncthreads();                                                       \
        p ^= 1;                                                                \
    }

__global__ __launch_bounds__(128) void gemm_f16_kernel(const _Float16* __restrict__ A,
                                                       const _Float16* __restrict__ W,
                                                       const float* __restrict__ bias,
                                                       _Float16* __restrict__ Out) {
    const int n0 = blockIdx.x * 64;
    const int m0 = blockIdx.y * 128;
    GEMM_BODY(A, W, n0, m0)

    const float b0 = bias[n0 +  0 + l15];
    const float b1 = bias[n0 + 16 + l15];
    const float b2 = bias[n0 + 32 + l15];
    const float b3 = bias[n0 + 48 + l15];
#pragma unroll
    for (int r = 0; r < 8; ++r) {                  // C layout: VGPR r -> M = r + sel*8
        const int mL = m0_w + r + sel * 8;
        _Float16* oL = Out + (size_t)mL * EMB + n0 + l15;
        oL[ 0] = (_Float16)(accL0[r] + b0);
        oL[16] = (_Float16)(accL1[r] + b1);
        oL[32] = (_Float16)(accL2[r] + b2);
        oL[48] = (_Float16)(accL3[r] + b3);
        _Float16* oH = oL + (size_t)16 * EMB;
        oH[ 0] = (_Float16)(accH0[r] + b0);
        oH[16] = (_Float16)(accH1[r] + b1);
        oH[32] = (_Float16)(accH2[r] + b2);
        oH[48] = (_Float16)(accH3[r] + b3);
    }
}

// Output projection with residual: out = x + A*W^T + bias (fp32 out)
__global__ __launch_bounds__(128) void gemm_out_kernel(const _Float16* __restrict__ A,
                                                       const _Float16* __restrict__ W,
                                                       const float* __restrict__ bias,
                                                       const float* __restrict__ xres,
                                                       float* __restrict__ Out) {
    const int n0 = blockIdx.x * 64;
    const int m0 = blockIdx.y * 128;
    GEMM_BODY(A, W, n0, m0)

    const float b0 = bias[n0 +  0 + l15];
    const float b1 = bias[n0 + 16 + l15];
    const float b2 = bias[n0 + 32 + l15];
    const float b3 = bias[n0 + 48 + l15];
#pragma unroll
    for (int r = 0; r < 8; ++r) {
        const int mL = m0_w + r + sel * 8;
        const float* xL = xres + (size_t)mL * EMB + n0 + l15;
        float* oL = Out + (size_t)mL * EMB + n0 + l15;
        oL[ 0] = accL0[r] + b0 + xL[ 0];
        oL[16] = accL1[r] + b1 + xL[16];
        oL[32] = accL2[r] + b2 + xL[32];
        oL[48] = accL3[r] + b3 + xL[48];
        const float* xH = xL + (size_t)16 * EMB;
        float* oH = oL + (size_t)16 * EMB;
        oH[ 0] = accH0[r] + b0 + xH[ 0];
        oH[16] = accH1[r] + b1 + xH[16];
        oH[32] = accH2[r] + b2 + xH[32];
        oH[48] = accH3[r] + b3 + xH[48];
    }
}

// ---------------------------------------------------------------------------
// V transpose: Vt[(b*H+h)*HD + d][key] = V[(b*SEQ+key)*EMB + h*HD + d]
// ---------------------------------------------------------------------------
__global__ __launch_bounds__(256) void transpose_v_kernel(const _Float16* __restrict__ V,
                                                          _Float16* __restrict__ Vt) {
    const int n = BATCH * HEADS * HD * SEQ;       // 4194304
    for (int idx = blockIdx.x * blockDim.x + threadIdx.x; idx < n;
         idx += gridDim.x * blockDim.x) {
        const int key = idx & (SEQ - 1);
        int rest      = idx >> 11;
        const int d   = rest & (HD - 1);
        rest >>= 6;
        const int h   = rest & (HEADS - 1);
        const int b   = rest >> 4;
        Vt[idx] = V[((size_t)(b * SEQ + key)) * EMB + h * HD + d];
    }
}

// ---------------------------------------------------------------------------
// Flash attention: one wave (32 threads) per 16-query tile per (b,h).
// Scores/PV via v_wmma_f32_16x16x32_f16; P staged C-layout -> A-layout in LDS.
// ---------------------------------------------------------------------------
__global__ __launch_bounds__(32) void attn_kernel(const _Float16* __restrict__ Q,
                                                  const _Float16* __restrict__ Kmat,
                                                  const _Float16* __restrict__ Vt,
                                                  _Float16* __restrict__ O) {
    const int lane = threadIdx.x & 31;
    const int l15  = lane & 15;
    const int sel  = lane >> 4;

    const int qt = blockIdx.x % (SEQ / 16);
    const int h  = (blockIdx.x / (SEQ / 16)) % HEADS;
    const int b  = blockIdx.x / ((SEQ / 16) * HEADS);
    const int q0 = qt * 16;
    const float scale = 0.125f;                   // 1/sqrt(64)

    __shared__ _Float16 lds_p[16 * 32];           // 16 q-rows x 32 keys

    const _Float16* qrow = Q + (size_t)(b * SEQ + q0 + l15) * EMB + h * HD;
    const v16h qa0 = load_fragA(qrow + 0,  sel);
    const v16h qa1 = load_fragA(qrow + 32, sel);

    v8f o0 = {}, o1 = {}, o2 = {}, o3 = {};
    float mi[8], li[8];
#pragma unroll
    for (int r = 0; r < 8; ++r) { mi[r] = -3.0e38f; li[r] = 0.f; }

    const _Float16* vbase = Vt + (size_t)(b * HEADS + h) * HD * SEQ;

    for (int j0 = 0; j0 < SEQ; j0 += 32) {
        // ---- scores S[16,32] = Q * K^T  (keys j0..j0+31) ----
        const _Float16* kr0 = Kmat + (size_t)(b * SEQ + j0 + l15) * EMB + h * HD;
        const _Float16* kr1 = kr0 + (size_t)16 * EMB;
        v8f s0 = {}, s1 = {};
        s0 = WMMA_F16(qa0, load_fragB(kr0 + 0,  sel), s0);
        s0 = WMMA_F16(qa1, load_fragB(kr0 + 32, sel), s0);
        s1 = WMMA_F16(qa0, load_fragB(kr1 + 0,  sel), s1);
        s1 = WMMA_F16(qa1, load_fragB(kr1 + 32, sel), s1);

        // ---- online softmax per row (row m = r + sel*8 lives in one 16-lane half) ----
#pragma unroll
        for (int r = 0; r < 8; ++r) {
            float a = s0[r] * scale;
            float c = s1[r] * scale;
            float mx = fmaxf(a, c);
            mx = fmaxf(mx, __shfl_xor(mx, 1, 32));
            mx = fmaxf(mx, __shfl_xor(mx, 2, 32));
            mx = fmaxf(mx, __shfl_xor(mx, 4, 32));
            mx = fmaxf(mx, __shfl_xor(mx, 8, 32));
            const float mnew  = fmaxf(mi[r], mx);
            const float alpha = __expf(mi[r] - mnew);
            const float p0 = __expf(a - mnew);
            const float p1 = __expf(c - mnew);
            float rsum = p0 + p1;
            rsum += __shfl_xor(rsum, 1, 32);
            rsum += __shfl_xor(rsum, 2, 32);
            rsum += __shfl_xor(rsum, 4, 32);
            rsum += __shfl_xor(rsum, 8, 32);
            li[r] = li[r] * alpha + rsum;
            mi[r] = mnew;
            o0[r] *= alpha; o1[r] *= alpha; o2[r] *= alpha; o3[r] *= alpha;

            const int m = r + sel * 8;            // C-layout row
            lds_p[m * 32 + l15]      = (_Float16)p0;
            lds_p[m * 32 + 16 + l15] = (_Float16)p1;
        }

        // ---- P (A-layout from LDS) times V (B from transposed V) ----
        const v16h pa = load_fragA(&lds_p[l15 * 32], sel);
        o0 = WMMA_F16(pa, load_fragB(vbase + (size_t)( 0 + l15) * SEQ + j0, sel), o0);
        o1 = WMMA_F16(pa, load_fragB(vbase + (size_t)(16 + l15) * SEQ + j0, sel), o1);
        o2 = WMMA_F16(pa, load_fragB(vbase + (size_t)(32 + l15) * SEQ + j0, sel), o2);
        o3 = WMMA_F16(pa, load_fragB(vbase + (size_t)(48 + l15) * SEQ + j0, sel), o3);
    }

#pragma unroll
    for (int r = 0; r < 8; ++r) {
        const int m = q0 + r + sel * 8;
        const float inv = 1.0f / li[r];
        _Float16* orow = O + (size_t)(b * SEQ + m) * EMB + h * HD + l15;
        orow[ 0] = (_Float16)(o0[r] * inv);
        orow[16] = (_Float16)(o1[r] * inv);
        orow[32] = (_Float16)(o2[r] * inv);
        orow[48] = (_Float16)(o3[r] * inv);
    }
}

// ---------------------------------------------------------------------------
// Host launcher
// ---------------------------------------------------------------------------
extern "C" void kernel_launch(void* const* d_in, const int* in_sizes, int n_in,
                              void* d_out, int out_size, void* d_ws, size_t ws_size,
                              hipStream_t stream) {
    (void)in_sizes; (void)n_in; (void)out_size; (void)ws_size;
    const float* x  = (const float*)d_in[0];
    const float* g  = (const float*)d_in[1];
    const float* be = (const float*)d_in[2];
    const float* Wq = (const float*)d_in[3];
    const float* bq = (const float*)d_in[4];
    const float* Wk = (const float*)d_in[5];
    const float* bk = (const float*)d_in[6];
    const float* Wv = (const float*)d_in[7];
    const float* bv = (const float*)d_in[8];
    const float* Wo = (const float*)d_in[9];
    const float* bo = (const float*)d_in[10];
    float* out = (float*)d_out;

    char* ws = (char*)d_ws;
    const size_t MiB = 1u << 20;
    _Float16* hB  = (_Float16*)(ws);               //  0 ..  8 MiB
    _Float16* WqF = (_Float16*)(ws +  8 * MiB);    //  8 .. 10
    _Float16* WkF = (_Float16*)(ws + 10 * MiB);    // 10 .. 12
    _Float16* WvF = (_Float16*)(ws + 12 * MiB);    // 12 .. 14
    _Float16* WoF = (_Float16*)(ws + 14 * MiB);    // 14 .. 16
    _Float16* Qf  = (_Float16*)(ws + 16 * MiB);    // 16 .. 24
    _Float16* Kf  = (_Float16*)(ws + 24 * MiB);    // 24 .. 32
    _Float16* Vf  = (_Float16*)(ws + 32 * MiB);    // 32 .. 40
    _Float16* VtF = (_Float16*)(ws + 40 * MiB);    // 40 .. 48
    _Float16* Of  = (_Float16*)(ws + 48 * MiB);    // 48 .. 56

    const int WN = EMB * EMB;                      // 1048576 per weight
    cast_f16_kernel<<<1024, 256, 0, stream>>>(Wq, WqF, WN);
    cast_f16_kernel<<<1024, 256, 0, stream>>>(Wk, WkF, WN);
    cast_f16_kernel<<<1024, 256, 0, stream>>>(Wv, WvF, WN);
    cast_f16_kernel<<<1024, 256, 0, stream>>>(Wo, WoF, WN);

    ln_kernel<<<MTOT, 256, 0, stream>>>(x, g, be, hB);

    dim3 gemm_grid(EMB / 64, MTOT / 128);          // (16, 32)
    gemm_f16_kernel<<<gemm_grid, 128, 0, stream>>>(hB, WqF, bq, Qf);
    gemm_f16_kernel<<<gemm_grid, 128, 0, stream>>>(hB, WkF, bk, Kf);
    gemm_f16_kernel<<<gemm_grid, 128, 0, stream>>>(hB, WvF, bv, Vf);

    transpose_v_kernel<<<4096, 256, 0, stream>>>(Vf, VtF);

    attn_kernel<<<BATCH * HEADS * (SEQ / 16), 32, 0, stream>>>(Qf, Kf, VtF, Of);

    gemm_out_kernel<<<gemm_grid, 128, 0, stream>>>(Of, WoF, bo, x, out);
}
